// FP8Linear_61452392071884
// MI455X (gfx1250) — 8-bit matrix operand order repaired and checked by reading, compiled, not run on hardware
//
#include <hip/hip_runtime.h>
#include <hip/hip_bf16.h>

typedef int   v16i __attribute__((ext_vector_type(16)));
typedef float v8f  __attribute__((ext_vector_type(8)));
typedef int   gv4i __attribute__((vector_size(16)));

#define FP8_MAX 448.0f
#define D 4096
#define M_TOTAL 8192

#define AS1 __attribute__((address_space(1)))
#define AS3 __attribute__((address_space(3)))

#if defined(__has_builtin)
#if __has_builtin(__builtin_amdgcn_global_load_async_to_lds_b128)
#define USE_ASYNC_LDS 1
#endif
#endif

__device__ __forceinline__ void wait_asynccnt0() {
#if defined(__has_builtin) && __has_builtin(__builtin_amdgcn_s_wait_asynccnt)
    __builtin_amdgcn_s_wait_asynccnt(0);
#else
    asm volatile("s_wait_asynccnt 0x0" ::: "memory");
#endif
}

__device__ __forceinline__ unsigned f32_to_e4m3_byte(float x) {
    unsigned sign = (__float_as_uint(x) >> 31) << 7;
    float ax = fminf(fabsf(x), FP8_MAX);
    unsigned au = __float_as_uint(ax);
    int e = (int)(au >> 23) - 127;
    if (e < -6) {
        int m = (int)rintf(ax * 512.0f);
        return sign | (unsigned)m;
    }
    float sc = __uint_as_float((unsigned)((127 + 3 - e) << 23));
    int m = (int)rintf(ax * sc);
    if (m == 16) { m = 8; e += 1; }
    if (e > 8) { e = 8; m = 14; }
    return sign | (unsigned)((e + 7) << 3) | (unsigned)(m & 7);
}

__device__ __forceinline__ unsigned pack4_e4m3(float f0, float f1, float f2, float f3) {
#if defined(__has_builtin) && __has_builtin(__builtin_amdgcn_cvt_pk_fp8_f32)
    f0 = fminf(fmaxf(f0, -FP8_MAX), FP8_MAX);
    f1 = fminf(fmaxf(f1, -FP8_MAX), FP8_MAX);
    f2 = fminf(fmaxf(f2, -FP8_MAX), FP8_MAX);
    f3 = fminf(fmaxf(f3, -FP8_MAX), FP8_MAX);
    int d = 0;
    d = __builtin_amdgcn_cvt_pk_fp8_f32(f0, f1, d, false);
    d = __builtin_amdgcn_cvt_pk_fp8_f32(f2, f3, d, true);
    return (unsigned)d;
#else
    return  f32_to_e4m3_byte(f0)
         | (f32_to_e4m3_byte(f1) << 8)
         | (f32_to_e4m3_byte(f2) << 16)
         | (f32_to_e4m3_byte(f3) << 24);
#endif
}

__global__ __launch_bounds__(256) void fwht_quant_kernel(
    const float* __restrict__ x, unsigned* __restrict__ xq_u32,
    float* __restrict__ sx_out)
{
    __shared__ float row[D];
    __shared__ float wmax[8];
    const int tid = threadIdx.x;
    const size_t r = blockIdx.x;
    const float* xr = x + r * (size_t)D;

    for (int i = tid; i < D; i += 256) row[i] = xr[i];
    __syncthreads();

    for (int s = 1; s < D; s <<= 1) {
        for (int j = tid; j < D / 2; j += 256) {
            int i = ((j & ~(s - 1)) << 1) | (j & (s - 1));
            float a = row[i], b = row[i + s];
            row[i]     = a + b;
            row[i + s] = a - b;
        }
        __syncthreads();
    }

    float mx = 0.0f;
    for (int i = tid; i < D; i += 256) {
        float v = row[i] * 0.015625f;
        row[i] = v;
        mx = fmaxf(mx, fabsf(v));
    }
    for (int off = 16; off > 0; off >>= 1)
        mx = fmaxf(mx, __shfl_xor(mx, off, 32));
    if ((tid & 31) == 0) wmax[tid >> 5] = mx;
    __syncthreads();
    if (tid == 0) {
        float m = wmax[0];
        for (int i = 1; i < 8; ++i) m = fmaxf(m, wmax[i]);
        m = fmaxf(m, 1e-20f);
        sx_out[r] = m / FP8_MAX;
        wmax[0] = FP8_MAX / m;
    }
    __syncthreads();
    const float inv = wmax[0];

    unsigned* outd = xq_u32 + r * (size_t)(D / 4);
    for (int d = tid; d < D / 4; d += 256) {
        int i = d * 4;
        outd[d] = pack4_e4m3(row[i] * inv, row[i + 1] * inv,
                             row[i + 2] * inv, row[i + 3] * inv);
    }
}

__global__ __launch_bounds__(256) void weight_quant_kernel(
    const float* __restrict__ w, unsigned* __restrict__ wq_u32)
{
    size_t d = (size_t)blockIdx.x * 256 + threadIdx.x;
    const float4 f = ((const float4*)w)[d];
    wq_u32[d] = pack4_e4m3(f.x, f.y, f.z, f.w);
}

__global__ __launch_bounds__(256) void gemm_fp8_kernel(
    const unsigned char* __restrict__ xq, const unsigned char* __restrict__ wq,
    const float* __restrict__ sx, const float* __restrict__ scales,
    const float* __restrict__ bias, float* __restrict__ out)
{
    __shared__ __align__(16) unsigned b_lds[2][4096];
    const int tid  = threadIdx.x;
    const int lane = tid & 31;
    const int wv   = tid >> 5;
    const int n0   = blockIdx.x * 128;
    const int m0   = blockIdx.y * 128;
    const int half = lane >> 4;
    const int l16  = lane & 15;

    const size_t mrow = (size_t)(m0 + wv * 16 + l16);
    const unsigned* arow = (const unsigned*)(xq + mrow * D);

    auto stage = [&](int buf, int k0) {
#pragma unroll
        for (int it = 0; it < 4; ++it) {
            int i = tid + it * 256;
            int rowb = i >> 3, c = i & 7;
#if defined(USE_ASYNC_LDS)
            __builtin_amdgcn_global_load_async_to_lds_b128(
                (AS1 gv4i*)(void*)(wq + (size_t)(n0 + rowb) * D + k0 + c * 16),
                (AS3 gv4i*)(void*)((char*)&b_lds[buf][0] + i * 16), 0, 0);
#else
            ((uint4*)&b_lds[buf][0])[i] =
                *(const uint4*)(wq + (size_t)(n0 + rowb) * D + k0 + c * 16);
#endif
        }
    };

    v8f acc[8];
#pragma unroll
    for (int nt = 0; nt < 8; ++nt)
        acc[nt] = (v8f){0.f, 0.f, 0.f, 0.f, 0.f, 0.f, 0.f, 0.f};

    stage(0, 0);
#if defined(USE_ASYNC_LDS)
    wait_asynccnt0();
#endif
    __syncthreads();

#pragma unroll 2
    for (int k0 = 0; k0 < D; k0 += 128) {
        const int cur = (k0 >> 7) & 1;
        const bool more = (k0 + 128 < D);
        if (more) stage(cur ^ 1, k0 + 128);

        v16i a;
#pragma unroll
        for (int v = 0; v < 16; ++v) {
            int kb = (v >> 2) * 32 + half * 16 + (v & 3) * 4;
            a[v] = arow[(k0 + kb) >> 2];
        }
        if (more)
            __builtin_prefetch((const char*)arow + k0 + 128, 0, 0);

#pragma unroll
        for (int nt = 0; nt < 8; ++nt) {
            const unsigned* brow = &b_lds[cur][(nt * 16 + l16) * 32];
            v16i b;
#pragma unroll
            for (int v = 0; v < 16; ++v) {
                int kb = (v >> 2) * 32 + half * 16 + (v & 3) * 4;
                b[v] = brow[kb >> 2];
            }
            acc[nt] = __builtin_amdgcn_wmma_f32_16x16x128_fp8_fp8(
                a, b, (short)0, acc[nt], false, false);
        }

        if (more) {
#if defined(USE_ASYNC_LDS)
            wait_asynccnt0();
#endif
            __syncthreads();
        }
    }

    const int mbase = m0 + wv * 16 + half * 8;
#pragma unroll
    for (int nt = 0; nt < 8; ++nt) {
        int ncol = n0 + nt * 16 + l16;
        float sc = scales[ncol];
        float bi = bias[ncol];
#pragma unroll
        for (int rr = 0; rr < 8; ++rr) {
            int rowo = mbase + rr;
            out[(size_t)rowo * D + ncol] = acc[nt][rr] * sx[rowo] * sc + bi;
        }
    }
}

extern "C" void kernel_launch(void* const* d_in, const int* in_sizes, int n_in,
                              void* d_out, int out_size, void* d_ws, size_t ws_size,
                              hipStream_t stream) {
    const float* x      = (const float*)d_in[0];
    const float* w      = (const float*)d_in[1];
    const float* scales = (const float*)d_in[2];
    const float* bias   = (const float*)d_in[3];
    float* out = (float*)d_out;

    unsigned char* xq = (unsigned char*)d_ws;
    unsigned char* wq = xq + (size_t)M_TOTAL * D;
    float* sx = (float*)(wq + (size_t)D * D);

    fwht_quant_kernel<<<M_TOTAL, 256, 0, stream>>>(x, (unsigned*)xq, sx);
    weight_quant_kernel<<<(D * (size_t)D / 4) / 256, 256, 0, stream>>>(
        w, (unsigned*)wq);
    dim3 grid(D / 128, M_TOTAL / 128);
    gemm_fp8_kernel<<<grid, 256, 0, stream>>>(xq, wq, sx, scales, bias, out);
}
